// mpqe_32203664785862
// MI455X (gfx1250) — matrix operand order repaired and checked by reading, not compiled, not run on hardware
//
#include <hip/hip_runtime.h>

// ---------------------------------------------------------------------------
// Problem constants (from reference)
// ---------------------------------------------------------------------------
#define N_LOCAL     16384
#define EMBED_DIM   256
#define NUM_REL     16
#define N_EDGES     131072
#define N_QUERIES   4096
#define K_TOTAL     (NUM_REL * EMBED_DIM + EMBED_DIM)   // 4352 = relation part + root part
#define K_TILES     (K_TOTAL / 32)                      // 136
#define K_REL_TILES ((NUM_REL * EMBED_DIM) / 32)        // 128

typedef unsigned short u16;
typedef unsigned int   u32;

typedef __attribute__((ext_vector_type(16))) __bf16 v16bf;
typedef __attribute__((ext_vector_type(8)))  float  v8f;

union BF16Frag {
    uint4 u4[2];
    v16bf v;
};

// ---------------------------------------------------------------------------
// 1) Gather: x[i,:] = node_emb[entity_ids[i], :]
// ---------------------------------------------------------------------------
__global__ void gather_rows_kernel(const float* __restrict__ emb,
                                   const int* __restrict__ ids,
                                   float* __restrict__ x) {
    const int i = blockIdx.x;
    const int c = threadIdx.x;
    x[(size_t)i * EMBED_DIM + c] = emb[(size_t)ids[i] * EMBED_DIM + c];
}

// ---------------------------------------------------------------------------
// 2) Build transposed bf16 weight: wT (256 x 4352), rows = output channel n,
//    cols k = r*256+d (relation weights) then 4096+d (root weight).
//    w flat index: ((r*256)+d)*256+n == k*256+n  for k<4096.
// ---------------------------------------------------------------------------
__global__ void build_wT_kernel(const float* __restrict__ w,     // (16,256,256)
                                const float* __restrict__ root,  // (256,256)
                                __bf16* __restrict__ wT) {       // (256,4352)
    const int idx = blockIdx.x * blockDim.x + threadIdx.x;
    if (idx >= EMBED_DIM * K_TOTAL) return;
    const int n = idx / K_TOTAL;
    const int k = idx % K_TOTAL;
    float v;
    if (k < NUM_REL * EMBED_DIM) v = w[(size_t)k * EMBED_DIM + n];
    else                         v = root[(size_t)(k - NUM_REL * EMBED_DIM) * EMBED_DIM + n];
    wT[idx] = (__bf16)v;
}

// ---------------------------------------------------------------------------
// 3) Edge scatter: agg[(tgt*16+rel), :] += x[src, :];  cnt[tgt*16+rel] += 1
// ---------------------------------------------------------------------------
__global__ void scatter_edges_kernel(const float* __restrict__ x,
                                     const int* __restrict__ src,
                                     const int* __restrict__ tgt,
                                     const int* __restrict__ rel,
                                     float* __restrict__ agg,
                                     float* __restrict__ cnt) {
    const int e = blockIdx.x;
    const int s  = src[e];
    const int tg = tgt[e];
    const int r  = rel[e];
    const int bucket = tg * NUM_REL + r;
    const int c = threadIdx.x;
    atomicAdd(&agg[(size_t)bucket * EMBED_DIM + c], x[(size_t)s * EMBED_DIM + c]);
    if (c == 0) atomicAdd(&cnt[bucket], 1.0f);
}

// ---------------------------------------------------------------------------
// 4) Fused RGCN GEMM (double-buffered, software-pipelined):
//    out(16384x256) = normalize(agg)(16384x4096) @ Wrel + x(16384x256) @ Wroot + b
//    Block = 256 threads (8 waves), C tile 128x128, wave tile 32x64,
//    K-step 32 -> 8x v_wmma_f32_16x16x32_bf16 per wave per step.
// ---------------------------------------------------------------------------
#define TM 128
#define TN 128
#define ASTR 40   // halves per LDS row (32 + 8 pad, keeps 16B row alignment)
#define BSTR 40

struct Stage {
    float4 f[4];   // A-tile fp32 values (16 elements)
    float  cntv;   // bucket count for this row/relation
    uint4  b0, b1; // B-tile bf16 values (16 elements)
};

// Global fetch for tile kt: issues loads early so they overlap WMMA work.
__device__ __forceinline__ void fetch_tile(int kt, int m0, int n0, int lm, int lk,
                                           const float* __restrict__ agg,
                                           const float* __restrict__ cnt,
                                           const float* __restrict__ x,
                                           const __bf16* __restrict__ wT,
                                           Stage& s) {
    if (kt < K_REL_TILES) {
        const int r   = kt >> 3;                 // relation for this k-tile
        const int kin = (kt & 7) * 32 + lk;      // offset within relation block
        s.cntv = cnt[(size_t)(m0 + lm) * NUM_REL + r];
        const float4* p = (const float4*)(agg +
            ((size_t)(m0 + lm) * NUM_REL + r) * EMBED_DIM + kin);
        s.f[0] = p[0]; s.f[1] = p[1]; s.f[2] = p[2]; s.f[3] = p[3];
    } else {
        const int kin = (kt - K_REL_TILES) * 32 + lk;   // root part: raw x
        s.cntv = 1.0f;
        const float4* p = (const float4*)(x + (size_t)(m0 + lm) * EMBED_DIM + kin);
        s.f[0] = p[0]; s.f[1] = p[1]; s.f[2] = p[2]; s.f[3] = p[3];
    }
    const __bf16* gsrc = wT + (size_t)(n0 + lm) * K_TOTAL + kt * 32 + lk;
    s.b0 = ((const uint4*)gsrc)[0];
    s.b1 = ((const uint4*)gsrc)[1];
    if (kt + 1 < K_TILES)
        __builtin_prefetch(gsrc + 32, 0, 3);     // global_prefetch_b8 next k-tile
}

// Convert + store staged tile into LDS buffer.
__device__ __forceinline__ void commit_tile(__bf16* __restrict__ A,
                                            __bf16* __restrict__ B,
                                            int lm, int lk, const Stage& s) {
    // cheap v_rcp_f32: feeds a bf16 cast, 1-ulp rcp is far below bf16 rounding
    const float scale = __builtin_amdgcn_rcpf(fmaxf(s.cntv, 1.0f));
    __bf16* dst = A + lm * ASTR + lk;
#pragma unroll
    for (int j = 0; j < 4; ++j) {
        dst[j * 4 + 0] = (__bf16)(s.f[j].x * scale);
        dst[j * 4 + 1] = (__bf16)(s.f[j].y * scale);
        dst[j * 4 + 2] = (__bf16)(s.f[j].z * scale);
        dst[j * 4 + 3] = (__bf16)(s.f[j].w * scale);
    }
    uint4* d = (uint4*)(B + lm * BSTR + lk);
    d[0] = s.b0;
    d[1] = s.b1;
}

__global__ void __launch_bounds__(256)
rgcn_gemm_kernel(const float* __restrict__ agg,   // (16384*16, 256)
                 const float* __restrict__ cnt,   // (16384*16)
                 const float* __restrict__ x,     // (16384, 256)
                 const __bf16* __restrict__ wT,   // (256, 4352) bf16, n-major
                 const float* __restrict__ bias,  // (256)
                 float* __restrict__ out) {       // (16384, 256)
    __shared__ __align__(16) __bf16 Alds[2][TM * ASTR];
    __shared__ __align__(16) __bf16 Blds[2][TN * BSTR];

    const int t    = threadIdx.x;
    const int lane = t & 31;
    const int wave = t >> 5;
    const int wm   = wave & 3;            // 4 M-groups of 32 rows
    const int wn   = wave >> 2;           // 2 N-groups of 64 cols
    const int m0   = blockIdx.x * TM;
    const int n0   = blockIdx.y * TN;

    const int fr    = lane & 15;          // fragment row/col within 16
    const int lhalf = (lane >> 4) & 1;    // lane group 0..15 vs 16..31

    v8f acc[2][4];
#pragma unroll
    for (int mi = 0; mi < 2; ++mi)
#pragma unroll
        for (int ni = 0; ni < 4; ++ni)
            acc[mi][ni] = v8f{};

    // staging assignment: thread t loads 16 elements of row (t>>1), half (t&1)
    const int lm = t >> 1;
    const int lk = (t & 1) * 16;

    // ---- pipeline prologue: stage tile 0 ----
    Stage st;
    fetch_tile(0, m0, n0, lm, lk, agg, cnt, x, wT, st);
    commit_tile(Alds[0], Blds[0], lm, lk, st);
    __syncthreads();

    for (int kt = 0; kt < K_TILES; ++kt) {
        const int cur = kt & 1;
        const int nxt = cur ^ 1;
        const bool has_next = (kt + 1) < K_TILES;

        // issue next tile's global loads first so they hide under the WMMAs
        if (has_next)
            fetch_tile(kt + 1, m0, n0, lm, lk, agg, cnt, x, wT, st);

        // ---- fragment loads (A and B both use contiguous k = 16*lhalf + 0..15) ----
        v16bf afrag[2], bfrag[4];
#pragma unroll
        for (int mi = 0; mi < 2; ++mi) {
            const __bf16* srcp = Alds[cur] + (wm * 32 + mi * 16 + fr) * ASTR;
            const int koffA = lhalf * 16;  // A: halves 0..15 -> K = 16*lhalf + 0..15
            BF16Frag u;
            u.u4[0] = *(const uint4*)(srcp + koffA);
            u.u4[1] = *(const uint4*)(srcp + 8 + koffA);
            afrag[mi] = u.v;
        }
#pragma unroll
        for (int ni = 0; ni < 4; ++ni) {
            const __bf16* srcp = Blds[cur] + (wn * 64 + ni * 16 + fr) * BSTR + lhalf * 16;
            BF16Frag u;                    // B: halves 0..15 -> K = 16*lhalf + 0..15
            u.u4[0] = ((const uint4*)srcp)[0];
            u.u4[1] = ((const uint4*)srcp)[1];
            bfrag[ni] = u.v;
        }

#pragma unroll
        for (int mi = 0; mi < 2; ++mi)
#pragma unroll
            for (int ni = 0; ni < 4; ++ni)
                acc[mi][ni] = __builtin_amdgcn_wmma_f32_16x16x32_bf16(
                    false, afrag[mi], false, bfrag[ni],
                    (short)0, acc[mi][ni], false, false);

        // write the buffer all waves already finished reading (barrier at kt-1)
        if (has_next)
            commit_tile(Alds[nxt], Blds[nxt], lm, lk, st);

        __syncthreads();
    }

    // ---- epilogue: C/D layout => lane n = fr, VGPR v row = v + 8*lhalf ----
#pragma unroll
    for (int mi = 0; mi < 2; ++mi) {
#pragma unroll
        for (int ni = 0; ni < 4; ++ni) {
            const int n = n0 + wn * 64 + ni * 16 + fr;
            const float bn = bias[n];
#pragma unroll
            for (int v = 0; v < 8; ++v) {
                const int m = m0 + wm * 32 + mi * 16 + lhalf * 8 + v;
                out[(size_t)m * EMBED_DIM + n] = acc[mi][ni][v] + bn;
            }
        }
    }
}

// ---------------------------------------------------------------------------
// 5) Readout: out[batch_ids[i], :] += x[i, :]
// ---------------------------------------------------------------------------
__global__ void readout_kernel(const float* __restrict__ x,
                               const int* __restrict__ bid,
                               float* __restrict__ out) {
    const int i = blockIdx.x;
    const int c = threadIdx.x;
    atomicAdd(&out[(size_t)bid[i] * EMBED_DIM + c], x[(size_t)i * EMBED_DIM + c]);
}

// ---------------------------------------------------------------------------
// Launch
// ---------------------------------------------------------------------------
extern "C" void kernel_launch(void* const* d_in, const int* in_sizes, int n_in,
                              void* d_out, int out_size, void* d_ws, size_t ws_size,
                              hipStream_t stream) {
    const float* node_emb  = (const float*)d_in[0];
    const float* w1        = (const float*)d_in[1];
    const float* root1     = (const float*)d_in[2];
    const float* b1        = (const float*)d_in[3];
    const float* w2        = (const float*)d_in[4];
    const float* root2     = (const float*)d_in[5];
    const float* b2        = (const float*)d_in[6];
    const int*   edge_idx  = (const int*)d_in[7];   // (2, E) flat
    const int*   edge_type = (const int*)d_in[8];
    const int*   ent_ids   = (const int*)d_in[9];
    const int*   batch_ids = (const int*)d_in[10];
    float*       outp      = (float*)d_out;

    // Workspace layout
    char* ws = (char*)d_ws;
    const size_t XB   = (size_t)N_LOCAL * EMBED_DIM * sizeof(float);           // 16 MB
    const size_t AGGB = (size_t)N_LOCAL * NUM_REL * EMBED_DIM * sizeof(float); // 64 MB
    const size_t CNTB = (size_t)N_LOCAL * NUM_REL * sizeof(float);             //  1 MB
    const size_t WTB  = (size_t)EMBED_DIM * K_TOTAL * sizeof(u16);             // 2.2 MB

    float*  x0  = (float*)ws;   ws += XB;
    float*  x1  = (float*)ws;   ws += XB;
    float*  agg = (float*)ws;   ws += AGGB;
    float*  cnt = (float*)ws;   ws += CNTB;
    __bf16* wt1 = (__bf16*)ws;  ws += WTB;
    __bf16* wt2 = (__bf16*)ws;  ws += WTB;

    const int* e_src = edge_idx;
    const int* e_tgt = edge_idx + N_EDGES;

    // Gather embeddings + prepare bf16 transposed weights
    gather_rows_kernel<<<N_LOCAL, EMBED_DIM, 0, stream>>>(node_emb, ent_ids, x0);
    {
        const int tot = EMBED_DIM * K_TOTAL;
        build_wT_kernel<<<(tot + 255) / 256, 256, 0, stream>>>(w1, root1, wt1);
        build_wT_kernel<<<(tot + 255) / 256, 256, 0, stream>>>(w2, root2, wt2);
    }

    dim3 gemm_grid(N_LOCAL / TM, EMBED_DIM / TN);

    // ---- Layer 1 ----
    hipMemsetAsync(agg, 0, AGGB, stream);
    hipMemsetAsync(cnt, 0, CNTB, stream);
    scatter_edges_kernel<<<N_EDGES, EMBED_DIM, 0, stream>>>(x0, e_src, e_tgt, edge_type, agg, cnt);
    rgcn_gemm_kernel<<<gemm_grid, 256, 0, stream>>>(agg, cnt, x0, wt1, b1, x1);

    // ---- Layer 2 ----
    hipMemsetAsync(agg, 0, AGGB, stream);
    hipMemsetAsync(cnt, 0, CNTB, stream);
    scatter_edges_kernel<<<N_EDGES, EMBED_DIM, 0, stream>>>(x1, e_src, e_tgt, edge_type, agg, cnt);
    rgcn_gemm_kernel<<<gemm_grid, 256, 0, stream>>>(agg, cnt, x1, wt2, b2, x0);

    // ---- Readout ----
    hipMemsetAsync(outp, 0, (size_t)N_QUERIES * EMBED_DIM * sizeof(float), stream);
    readout_kernel<<<N_LOCAL, EMBED_DIM, 0, stream>>>(x0, batch_ids, outp);
}
